// S6_10986526343391
// MI455X (gfx1250) — matrix operand order repaired and checked by reading, not compiled, not run on hardware
//
#include <hip/hip_runtime.h>
#include <hip/hip_bf16.h>
#include <cstdint>
#include <cstddef>

#define D_DIM 1024
#define N_DIM 64
#define EPSV  1e-6f

typedef __attribute__((ext_vector_type(8)))  float          v8f;
typedef __attribute__((ext_vector_type(16))) __bf16         v16bf;
typedef __attribute__((ext_vector_type(16))) unsigned short v16u;
typedef __attribute__((ext_vector_type(8)))  unsigned short v8u;

static __device__ __forceinline__ unsigned short f2bf(float f) {
    union { float f; unsigned u; } v; v.f = f;
    unsigned u = v.u;
    unsigned r = u + 0x7FFFu + ((u >> 16) & 1u);   // round-to-nearest-even
    return (unsigned short)(r >> 16);
}

static __device__ __forceinline__ float softplus_f(float v) {
    return (v > 20.0f) ? v : log1pf(expf(v));
}

// ---------------------------------------------------------------------------
// K0: weight conversion, padded Wd tile, A = -softplus(A_log)
// ---------------------------------------------------------------------------
__global__ __launch_bounds__(256) void prep_kernel(
    const float* __restrict__ WB, const float* __restrict__ Wo,
    const float* __restrict__ Wd, const float* __restrict__ A_log,
    unsigned short* __restrict__ WBb, unsigned short* __restrict__ Wob,
    unsigned short* __restrict__ Wpad, float* __restrict__ Aneg) {
    int i = blockIdx.x * blockDim.x + threadIdx.x;
    if (i < N_DIM * D_DIM) {
        WBb[i] = f2bf(WB[i]);
        Wob[i] = f2bf(Wo[i]);
    }
    if (i < 16 * D_DIM)                      // row 0 = Wd, rows 1..15 = 0
        Wpad[i] = (i < D_DIM) ? f2bf(Wd[i]) : (unsigned short)0;
    if (i < N_DIM) Aneg[i] = -softplus_f(A_log[i]);
}

// ---------------------------------------------------------------------------
// K1: B_t = x@WB^T (+bB) and Delta-linear via a 5th WMMA against the padded
//     Wd tile (column 0). Emits dA, dB transposed (B, N, L) for the scan.
//     One wave = one 16-row M-tile, all 64 n columns.
// ---------------------------------------------------------------------------
__global__ __launch_bounds__(256) void proj_kernel(
    const float* __restrict__ x, const unsigned short* __restrict__ WBb,
    const unsigned short* __restrict__ Wpad, const float* __restrict__ bdp,
    const float* __restrict__ bB, const float* __restrict__ Aneg,
    float* __restrict__ dAt, float* __restrict__ dBt) {
    __shared__ float sdelta[8][16];
    const int lane = threadIdx.x & 31;
    const int warp = threadIdx.x >> 5;
    const int Mtile = blockIdx.x * 8 + warp;       // 0..2047
    const int sel = lane >> 4;                     // half-wave select
    const int mrow = lane & 15;
    const int row0 = Mtile * 16;
    const float* xrow = x + (size_t)(row0 + mrow) * D_DIM;

    v8f cacc[5] = {};

    for (int kb = 0; kb < D_DIM; kb += 32) {
        const int off1 = kb + sel * 16;            // K {0..7} / {16..23}
        const int off2 = kb + sel * 16 + 8;        // K {8..15} / {24..31}
        float4 xa = *(const float4*)(xrow + off1);
        float4 xb = *(const float4*)(xrow + off1 + 4);
        float4 xc = *(const float4*)(xrow + off2);
        float4 xd = *(const float4*)(xrow + off2 + 4);

        __builtin_prefetch(xrow + kb + 256, 0, 1); // stream x ~8 blocks ahead

        v16bf af;
        af[0]  = (__bf16)xa.x; af[1]  = (__bf16)xa.y;
        af[2]  = (__bf16)xa.z; af[3]  = (__bf16)xa.w;
        af[4]  = (__bf16)xb.x; af[5]  = (__bf16)xb.y;
        af[6]  = (__bf16)xb.z; af[7]  = (__bf16)xb.w;
        af[8]  = (__bf16)xc.x; af[9]  = (__bf16)xc.y;
        af[10] = (__bf16)xc.z; af[11] = (__bf16)xc.w;
        af[12] = (__bf16)xd.x; af[13] = (__bf16)xd.y;
        af[14] = (__bf16)xd.z; af[15] = (__bf16)xd.w;

        const int koff = kb + sel * 16;            // B frag: 16 contiguous K
        #pragma unroll
        for (int nt = 0; nt < 4; ++nt) {
            const unsigned short* bp =
                WBb + (size_t)(nt * 16 + mrow) * D_DIM + koff;
            v16bf bfrag = __builtin_bit_cast(v16bf, *(const v16u*)bp);
            cacc[nt] = __builtin_amdgcn_wmma_f32_16x16x32_bf16(
                false, af, false, bfrag, (short)0, cacc[nt], false, false);
        }
        {   // Delta projection: column 0 of this tile is x @ Wd^T
            const unsigned short* dp = Wpad + (size_t)mrow * D_DIM + koff;
            v16bf dfrag = __builtin_bit_cast(v16bf, *(const v16u*)dp);
            cacc[4] = __builtin_amdgcn_wmma_f32_16x16x32_bf16(
                false, af, false, dfrag, (short)0, cacc[4], false, false);
        }
    }

    // Delta-linear per row lives on lanes 0 (rows 0-7) and 16 (rows 8-15).
    if (mrow == 0) {
        #pragma unroll
        for (int j = 0; j < 8; ++j) sdelta[warp][j + sel * 8] = cacc[4][j];
    }
    __syncthreads();
    const float delta = softplus_f(sdelta[warp][mrow] + bdp[0]);

    const int b  = Mtile >> 8;                     // 256 tiles per batch
    const int l0 = (Mtile & 255) * 16;

    #pragma unroll
    for (int nt = 0; nt < 4; ++nt) {
        const int n = nt * 16 + mrow;
        const float An  = Aneg[n];
        const float bBn = bB[n];
        const float inv = 1.0f / (An + EPSV);
        float da8[8], db8[8];
        #pragma unroll
        for (int j = 0; j < 8; ++j) {
            const int m = j + sel * 8;             // C layout row
            const float Dm = __shfl(delta, m);
            const float da = expf(An * Dm);
            const float bt = cacc[nt][j] + bBn;
            da8[j] = da;
            db8[j] = (da - 1.0f) * inv * bt;
        }
        const size_t base = ((size_t)(b * N_DIM + n)) * 4096 + l0 + sel * 8;
        *(float4*)(dAt + base)     = make_float4(da8[0], da8[1], da8[2], da8[3]);
        *(float4*)(dAt + base + 4) = make_float4(da8[4], da8[5], da8[6], da8[7]);
        *(float4*)(dBt + base)     = make_float4(db8[0], db8[1], db8[2], db8[3]);
        *(float4*)(dBt + base + 4) = make_float4(db8[4], db8[5], db8[6], db8[7]);
    }
}

// ---------------------------------------------------------------------------
// K2: per-(b,n) scan along L. One wave per scan; Hillis-Steele cross-lane
//     scans over 128 chunks of 32 with running prefix product/sum.
//     Emits G = Ct^2 * s as bf16 in (row, n) layout (A-operand for K3).
// ---------------------------------------------------------------------------
__global__ __launch_bounds__(256) void scan_kernel(
    const float* __restrict__ dAt, const float* __restrict__ dBt,
    unsigned short* __restrict__ Gb) {
    const int lane = threadIdx.x & 31;
    const int w = (blockIdx.x * blockDim.x + threadIdx.x) >> 5;  // 0..511
    const int b = w >> 6;
    const int n = w & 63;
    const float* pa = dAt + (size_t)w * 4096;
    const float* pd = dBt + (size_t)w * 4096;

    float P = 1.0f;   // running prefix product of dA
    float S = 0.0f;   // running prefix sum of z
    for (int c = 0; c < 128; ++c) {
        const int l = c * 32 + lane;
        const float a  = pa[l];
        const float dv = pd[l];

        float ia = a;                              // inclusive product scan
        #pragma unroll
        for (int s = 1; s < 32; s <<= 1) {
            float t = __shfl_up(ia, s);
            if (lane >= s) ia *= t;
        }
        float ex = __shfl_up(ia, 1);
        if (lane == 0) ex = 1.0f;                  // exclusive product
        const float Ct = P * ex;                   // shifted cumprod
        const float z = dv / (Ct + EPSV);

        float iz = z;                              // inclusive sum scan
        #pragma unroll
        for (int s = 1; s < 32; s <<= 1) {
            float t = __shfl_up(iz, s);
            if (lane >= s) iz += t;
        }
        const float sv = S + iz;                   // global cumsum
        const float G = Ct * Ct * sv;              // Ct * X_state
        Gb[((size_t)(b * 4096 + l)) * N_DIM + n] = f2bf(G);

        P *= __shfl(ia, 31);
        S  = __shfl(sv, 31);
    }
}

// ---------------------------------------------------------------------------
// K3: Y = G @ Wo^T + bo. One wave = 16(l) x 64(d) tile, K=64 in 2 WMMA steps
//     per 16-col tile. LDS transpose for contiguous 256B row-segment stores.
// ---------------------------------------------------------------------------
__global__ __launch_bounds__(256) void out_kernel(
    const unsigned short* __restrict__ Gb, const unsigned short* __restrict__ Wob,
    const float* __restrict__ bo, float* __restrict__ out) {
    __shared__ float lds[8 * 16 * 64];             // 32 KB
    const int lane = threadIdx.x & 31;
    const int warp = threadIdx.x >> 5;
    const int t = blockIdx.x * 8 + warp;           // 0..32767 wave-tiles
    const int Mtile = t >> 4;
    const int d0 = (t & 15) * 64;
    const int sel = lane >> 4;
    const int mrow = lane & 15;
    const int row0 = Mtile * 16;

    struct V16S { v8u lo, hi; };
    const unsigned short* gr = Gb + (size_t)(row0 + mrow) * N_DIM;
    V16S s0 = { *(const v8u*)(gr + 0  + sel * 16), *(const v8u*)(gr + 8  + sel * 16) };
    V16S s1 = { *(const v8u*)(gr + 32 + sel * 16), *(const v8u*)(gr + 40 + sel * 16) };
    v16bf a0 = __builtin_bit_cast(v16bf, s0);
    v16bf a1 = __builtin_bit_cast(v16bf, s1);

    v8f cacc[4] = {};
    #pragma unroll
    for (int ct = 0; ct < 4; ++ct) {
        const unsigned short* wr =
            Wob + (size_t)(d0 + ct * 16 + mrow) * N_DIM + sel * 16;
        v16bf b0 = __builtin_bit_cast(v16bf, *(const v16u*)(wr));
        v16bf b1 = __builtin_bit_cast(v16bf, *(const v16u*)(wr + 32));
        cacc[ct] = __builtin_amdgcn_wmma_f32_16x16x32_bf16(
            false, a0, false, b0, (short)0, cacc[ct], false, false);
        cacc[ct] = __builtin_amdgcn_wmma_f32_16x16x32_bf16(
            false, a1, false, b1, (short)0, cacc[ct], false, false);
    }

    float* tile = lds + warp * 1024;
    #pragma unroll
    for (int ct = 0; ct < 4; ++ct)
        #pragma unroll
        for (int j = 0; j < 8; ++j)
            tile[(j + sel * 8) * 64 + ct * 16 + mrow] = cacc[ct][j];
    __syncthreads();

    #pragma unroll
    for (int it = 0; it < 8; ++it) {
        const int idx = it * 32 + lane;            // 0..255 float4s
        const int r  = idx >> 4;
        const int c4 = (idx & 15) * 4;
        float4 v  = *(float4*)(tile + r * 64 + c4);
        float4 bb = *(const float4*)(bo + d0 + c4);
        v.x += bb.x; v.y += bb.y; v.z += bb.z; v.w += bb.w;
        *(float4*)(out + (size_t)(row0 + r) * D_DIM + d0 + c4) = v;
    }
}

// ---------------------------------------------------------------------------
extern "C" void kernel_launch(void* const* d_in, const int* in_sizes, int n_in,
                              void* d_out, int out_size, void* d_ws, size_t ws_size,
                              hipStream_t stream) {
    const float* x     = (const float*)d_in[0];
    const float* A_log = (const float*)d_in[1];
    const float* WB    = (const float*)d_in[2];
    const float* bB    = (const float*)d_in[3];
    // d_in[4] = WC, d_in[5] = bC : dead in the reference (overwritten)
    const float* Wd    = (const float*)d_in[6];
    const float* bd    = (const float*)d_in[7];
    const float* Wo    = (const float*)d_in[8];
    const float* bo    = (const float*)d_in[9];
    float* out = (float*)d_out;

    char* ws = (char*)d_ws;
    unsigned short* WBb  = (unsigned short*)(ws);                      // 128 KB
    unsigned short* Wob  = (unsigned short*)(ws + 131072);             // 128 KB
    float* Aneg          = (float*)(ws + 262144);                      // 256 B
    unsigned short* Wpad = (unsigned short*)(ws + 262400);             // 32 KB
    float* dAt           = (float*)(ws + 295168);                      // 8 MB
    float* dBt           = (float*)(ws + 295168 + 8388608);            // 8 MB
    unsigned short* Gb   = (unsigned short*)(ws + 295168 + 16777216);  // 4 MB

    prep_kernel<<<256, 256, 0, stream>>>(WB, Wo, Wd, A_log, WBb, Wob, Wpad, Aneg);
    proj_kernel<<<256, 256, 0, stream>>>(x, WBb, Wpad, bd, bB, Aneg, dAt, dBt);
    scan_kernel<<<64, 256, 0, stream>>>(dAt, dBt, Gb);
    out_kernel<<<4096, 256, 0, stream>>>(Gb, Wob, bo, out);
}
